// AdaptiveLeaf_40381282517136
// MI455X (gfx1250) — matrix operand order repaired and checked by reading, not compiled, not run on hardware
//
#include <hip/hip_runtime.h>
#include <hip/hip_bf16.h>
#include <math.h>

typedef __attribute__((ext_vector_type(16))) _Float16 v16h;
typedef __attribute__((ext_vector_type(8)))  float    v8f;

#define NF      40        // mel filters
#define KF      401       // filter taps
#define KPAD    416       // taps padded to 13*32
#define NKC     13        // K chunks of 32
#define NMT     5         // M tiles: 80 rows (re/im interleaved) / 16
#define LSIG    160000
#define NB      8
#define NT      1000
#define STRIDE  160
#define PADC    200
#define HID     32

#define SPAN_POS 2816     // 176 N-tiles of 16 positions per frame-tile
#define SEG_LEN  3232     // SPAN_POS + 400 halo, padded

// workspace byte offsets
#define OFF_FILT   0u                       // _Float16 [5][13][32][16]  (A-fragment layout)
#define OFF_WIN    66560u                   // _Float16 [40][416]
#define OFF_FRAMES 99840u                   // float [8*40*1000]
#define OFF_ALPHA  (OFF_FRAMES + 1280000u)  // float [320000]
#define OFF_R      (OFF_ALPHA  + 1280000u)  // float [320000]

__device__ __forceinline__ float sigmf(float x) { return 1.f / (1.f + __expf(-x)); }

// ---------------------------------------------------------------------------
// Kernel 1: Gabor filter coefficients, written directly in WMMA A-fragment
// layout.  Row order within each 16-row M-tile: [f0_re, f0_im, f1_re, ...]
// so re/im pairs land in adjacent accumulator VGPRs of the same lane.
// A 16x32 f16 layout: lanes 0-15 row=l, K={0..7,16..23}; lanes 16-31 row=l-16,
// K={8..15,24..31}.
// ---------------------------------------------------------------------------
__global__ void k_filt(const float* __restrict__ cfi, const float* __restrict__ bwi,
                       _Float16* __restrict__ filt) {
  int mt = blockIdx.x / NKC, kc = blockIdx.x % NKC;
  int lane = threadIdx.x;
  const float zc = 0.374834150f;            // sqrt(2 ln2)/pi
  _Float16* dst = filt + ((size_t)((mt * NKC) + kc) * 32 + lane) * 16;
  int row = (lane < 16) ? lane : lane - 16;
  int f = mt * 8 + (row >> 1);
  bool re = ((row & 1) == 0);
  float cf = fminf(fmaxf(cfi[f], 0.f), 3.14159265f);
  float bw = fminf(fmaxf(bwi[f], 4.f * zc), (float)KF * zc);
  float denom = 1.f / (sqrtf(2.f * 3.14159265f) * bw);
  float inv2b2 = 1.f / (2.f * bw * bw);
#pragma unroll
  for (int j = 0; j < 16; ++j) {
    int kk = kc * 32 + ((j < 8) ? j : j + 8) + ((lane >= 16) ? 8 : 0);
    float v = 0.f;
    if (kk < KF) {
      float t = (float)(kk - PADC);
      float g = denom * __expf(-t * t * inv2b2);
      float ph = cf * t;
      v = g * (re ? __cosf(ph) : __sinf(ph));
    }
    dst[j] = (_Float16)v;
  }
}

// ---------------------------------------------------------------------------
// Kernel 2: Gaussian pooling windows (f16, padded to 416)
// ---------------------------------------------------------------------------
__global__ void k_win(const float* __restrict__ pwi, _Float16* __restrict__ win) {
  int f = blockIdx.x;
  float pw = fminf(fmaxf(pwi[f], 2.0f / KF), 0.5f);
  for (int u = threadIdx.x; u < KPAD; u += blockDim.x) {
    float v = 0.f;
    if (u < KF) {
      float tt = (float)u / (KF - 1) - 0.5f;
      float q = tt / pw;
      v = __expf(-0.5f * q * q);
    }
    win[f * KPAD + u] = (_Float16)v;
  }
}

// ---------------------------------------------------------------------------
// Kernel 3: fused Gabor conv (WMMA im2col GEMM) + |.|^2 + Gaussian pooling.
// Block = (frame-tile of 16 output frames, batch).  8 waves; each wave owns
// 22 of the 176 position N-tiles.  K = 401 taps -> 13 chunks of 32.
// ---------------------------------------------------------------------------
__global__ void k_conv(const float* __restrict__ x, const _Float16* __restrict__ filt,
                       const _Float16* __restrict__ winw, float* __restrict__ frames) {
  extern __shared__ char smem[];
  _Float16* xseg = (_Float16*)smem;                            // [3232] f16 signal segment
  _Float16* win  = (_Float16*)(smem + SEG_LEN * 2);            // [40][416] pooling windows
  float* acc     = (float*)(smem + SEG_LEN * 2 + NF * KPAD * 2); // [40][16] frame accum

  int t0 = blockIdx.x * 16;
  int b = blockIdx.y;
  int tid = threadIdx.x;
  int seg0 = t0 * STRIDE - 2 * PADC;

  for (int i = tid; i < SEG_LEN; i += blockDim.x) {
    int g = seg0 + i;
    float v = (g >= 0 && g < LSIG) ? x[(size_t)b * LSIG + g] : 0.f;
    xseg[i] = (_Float16)v;
  }
  for (int i = tid; i < NF * KPAD; i += blockDim.x) win[i] = winw[i];
  for (int i = tid; i < NF * 16; i += blockDim.x) acc[i] = 0.f;
  __syncthreads();

  int wave = tid >> 5, lane = tid & 31;
  int col = lane & 15;
  v8f zero = {};
  for (int nt = wave; nt < SPAN_POS / 16; nt += 8) {   // 176/8 = 22 per wave, uniform
    int n0 = nt * 16;
    v8f d[NMT];
#pragma unroll
    for (int m = 0; m < NMT; ++m) d[m] = zero;
    for (int kc = 0; kc < NKC; ++kc) {
      // B 32x16 f16: lanes 0-15 col=l, K={0..7,16..23}; lanes 16-31 col=l-16, K={8..15,24..31}
      v16h bf;
#pragma unroll
      for (int j = 0; j < 16; ++j) {
        int kk = kc * 32 + ((j < 8) ? j : j + 8) + ((lane >= 16) ? 8 : 0);
        bf[j] = xseg[n0 + col + kk];
      }
#pragma unroll
      for (int m = 0; m < NMT; ++m) {
        v16h af = *(const v16h*)(filt + ((size_t)((m * NKC) + kc) * 32 + lane) * 16);
        d[m] = __builtin_amdgcn_wmma_f32_16x16x32_f16(false, af, false, bf,
                                                      (short)0, d[m], false, false);
      }
    }
    // squared modulus (re/im adjacent in D VGPRs) + pooled accumulation
    int pos = t0 * STRIDE - PADC + n0 + col;
    if (pos >= 0 && pos < LSIG) {
      int dlo = pos - PADC;
      int tlo = (dlo <= 0) ? 0 : (dlo + STRIDE - 1) / STRIDE;
      if (tlo < t0) tlo = t0;
      int thi = (pos + PADC) / STRIDE;
      if (thi > t0 + 15) thi = t0 + 15;
      if (thi > NT - 1) thi = NT - 1;
#pragma unroll
      for (int m = 0; m < NMT; ++m) {
#pragma unroll
        for (int i = 0; i < 4; ++i) {
          float re = d[m][2 * i], im = d[m][2 * i + 1];
          float mod = re * re + im * im;
          int f = m * 8 + ((lane >= 16) ? 4 : 0) + i;
          for (int t = tlo; t <= thi; ++t) {
            int u = pos - t * STRIDE + PADC;
            atomicAdd(&acc[f * 16 + (t - t0)], mod * (float)win[f * KPAD + u]);
          }
        }
      }
    }
  }
  __syncthreads();
  for (int i = tid; i < NF * 16; i += blockDim.x) {
    int f = i >> 4, j = i & 15, t = t0 + j;
    if (t < NT) frames[((size_t)b * NF + f) * NT + t] = acc[i];
  }
}

// ---------------------------------------------------------------------------
// Kernel 4: PCEN controller, fully parallel over (b,f,t) since the GRU/MLP
// depend only on (X_{t-1}, X_t).  Cell 1 collapses (h=0, biases known) to a
// per-lane elementwise pass; cell 2's h1*Whh^T and the head matmul run as
// WMMA with weight B-fragments held persistently in VGPRs.
// ---------------------------------------------------------------------------
__global__ void k_ctrl(const float* __restrict__ frames,
                       const float* __restrict__ w_ih, const float* __restrict__ w_hh,
                       const float* __restrict__ b_ih, const float* __restrict__ b_hh,
                       const float* __restrict__ hw1, const float* __restrict__ hb1,
                       const float* __restrict__ hw2, const float* __restrict__ hb2,
                       float* __restrict__ alpha_o, float* __restrict__ r_o) {
  __shared__ float s_wih[96], s_bih[96], s_bhh[96], s_hb1[32], s_hw2[64], s_hb2[2];
  __shared__ _Float16 h1s[8][32][32];   // per-wave h1 (then h2) staging
  __shared__ float xts[8][32];
  __shared__ float hids[8][16][32];

  int tid = threadIdx.x, wave = tid >> 5, lane = tid & 31;
  if (tid < 96) { s_wih[tid] = w_ih[tid]; s_bih[tid] = b_ih[tid]; s_bhh[tid] = b_hh[tid]; }
  if (tid < 32) s_hb1[tid] = hb1[tid];
  if (tid < 64) s_hw2[tid] = hw2[tid];
  if (tid < 2)  s_hb2[tid] = hb2[tid];

  // persistent weight B-fragments (K=32 == HID exactly): gh tiles 0..5, head tiles 0..1
  int col = lane & 15;
  int khi = (lane >= 16) ? 8 : 0;
  v16h Bg[6], Bh[2];
#pragma unroll
  for (int n = 0; n < 6; ++n)
#pragma unroll
    for (int j = 0; j < 16; ++j)
      Bg[n][j] = (_Float16)w_hh[(n * 16 + col) * HID + (((j < 8) ? j : j + 8) + khi)];
#pragma unroll
  for (int n = 0; n < 2; ++n)
#pragma unroll
    for (int j = 0; j < 16; ++j)
      Bh[n][j] = (_Float16)hw1[(n * 16 + col) * HID + (((j < 8) ? j : j + 8) + khi)];
  __syncthreads();

  int inst = blockIdx.x * 256 + tid;     // (b*40+f)*1000 + t
  int t = inst % NT;
  float xt = frames[inst];
  float xp = (t > 0) ? frames[inst - 1] : xt;
  xts[wave][lane] = xt;
  // GRU cell 1 with h=0: gh == b_hh, purely elementwise
#pragma unroll 4
  for (int j = 0; j < HID; ++j) {
    float r = sigmf(xp * s_wih[j] + s_bih[j] + s_bhh[j]);
    float z = sigmf(xp * s_wih[32 + j] + s_bih[32 + j] + s_bhh[32 + j]);
    float nn = tanhf(xp * s_wih[64 + j] + s_bih[64 + j] + r * s_bhh[64 + j]);
    h1s[wave][lane][j] = (_Float16)((1.f - z) * nn);
  }
  __syncthreads();

  int rowbase = (lane >= 16) ? 8 : 0;
  v8f zero = {};
  for (int g = 0; g < 2; ++g) {          // two 16-instance row-groups per wave
    int arow = g * 16 + ((lane < 16) ? lane : lane - 16);
    v16h af;
#pragma unroll
    for (int j = 0; j < 16; ++j) {
      int kk = ((j < 8) ? j : j + 8) + ((lane >= 16) ? 8 : 0);
      af[j] = h1s[wave][arow][kk];
    }
    v8f d[6];
#pragma unroll
    for (int n = 0; n < 6; ++n)
      d[n] = __builtin_amdgcn_wmma_f32_16x16x32_f16(false, af, false, Bg[n],
                                                    (short)0, zero, false, false);
    // GRU cell 2 gates (D layout: lane = col, VGPR = row)
#pragma unroll
    for (int v = 0; v < 8; ++v) {
      int il = g * 16 + rowbase + v;
      float xv = xts[wave][il];
#pragma unroll
      for (int hh = 0; hh < 2; ++hh) {
        int j = hh * 16 + col;
        float r = sigmf(xv * s_wih[j] + s_bih[j] + d[hh][v]);
        float z = sigmf(xv * s_wih[32 + j] + s_bih[32 + j] + d[2 + hh][v]);
        float nn = tanhf(xv * s_wih[64 + j] + s_bih[64 + j] + r * d[4 + hh][v]);
        float h1v = (float)h1s[wave][il][j];
        h1s[wave][il][j] = (_Float16)((1.f - z) * nn + z * h1v);
      }
    }
    __syncthreads();
    // head: relu(h2 @ hw1^T + hb1)
    v16h af2;
#pragma unroll
    for (int j = 0; j < 16; ++j) {
      int kk = ((j < 8) ? j : j + 8) + ((lane >= 16) ? 8 : 0);
      af2[j] = h1s[wave][arow][kk];
    }
    v8f e0 = __builtin_amdgcn_wmma_f32_16x16x32_f16(false, af2, false, Bh[0],
                                                    (short)0, zero, false, false);
    v8f e1 = __builtin_amdgcn_wmma_f32_16x16x32_f16(false, af2, false, Bh[1],
                                                    (short)0, zero, false, false);
#pragma unroll
    for (int v = 0; v < 8; ++v) {
      int r16 = rowbase + v;
      hids[wave][r16][col]      = fmaxf(e0[v] + s_hb1[col], 0.f);
      hids[wave][r16][16 + col] = fmaxf(e1[v] + s_hb1[16 + col], 0.f);
    }
    __syncthreads();
    // logits: 2 outputs -> plain VALU dot; lane = (logit, instance)
    int li = lane >> 4;
    int ii = lane & 15;
    float a2 = s_hb2[li];
#pragma unroll 8
    for (int j = 0; j < HID; ++j) a2 += hids[wave][ii][j] * s_hw2[li * HID + j];
    int gi = blockIdx.x * 256 + wave * 32 + g * 16 + ii;
    if (li == 0) alpha_o[gi] = sigmf(a2);
    else         r_o[gi] = 0.2f + 0.8f * sigmf(a2);
    __syncthreads();
  }
}

// ---------------------------------------------------------------------------
// Kernel 5: sequential PCEN IIR + output nonlinearity (320 independent bins)
// ---------------------------------------------------------------------------
__global__ void k_pcen(const float* __restrict__ frames, const float* __restrict__ al,
                       const float* __restrict__ rr, float* __restrict__ out) {
  int bin = blockIdx.x * blockDim.x + threadIdx.x;
  if (bin >= NB * NF) return;
  size_t base = (size_t)bin * NT;
  float M = 0.f;
  for (int t = 0; t < NT; ++t) {
    float X = frames[base + t];
    M = 0.96f * M + 0.04f * X;
    float a = al[base + t];
    float rv = rr[base + t];
    float den = __powf(M + 1e-6f, a);
    out[base + t] = __powf(X / den + 2.f, rv) - __powf(2.f, rv);
  }
}

extern "C" void kernel_launch(void* const* d_in, const int* in_sizes, int n_in,
                              void* d_out, int out_size, void* d_ws, size_t ws_size,
                              hipStream_t stream) {
  (void)in_sizes; (void)n_in; (void)out_size; (void)ws_size;
  const float* x   = (const float*)d_in[0];
  const float* cf  = (const float*)d_in[1];
  const float* bw  = (const float*)d_in[2];
  const float* pw  = (const float*)d_in[3];
  const float* wih = (const float*)d_in[4];
  const float* whh = (const float*)d_in[5];
  const float* bih = (const float*)d_in[6];
  const float* bhh = (const float*)d_in[7];
  const float* hw1 = (const float*)d_in[8];
  const float* hb1 = (const float*)d_in[9];
  const float* hw2 = (const float*)d_in[10];
  const float* hb2 = (const float*)d_in[11];

  char* ws = (char*)d_ws;
  _Float16* filt = (_Float16*)(ws + OFF_FILT);
  _Float16* win  = (_Float16*)(ws + OFF_WIN);
  float* frames  = (float*)(ws + OFF_FRAMES);
  float* al      = (float*)(ws + OFF_ALPHA);
  float* rr      = (float*)(ws + OFF_R);
  float* out     = (float*)d_out;

  k_filt<<<NMT * NKC, 32, 0, stream>>>(cf, bw, filt);
  k_win<<<NF, 256, 0, stream>>>(pw, win);
  size_t ldsB = SEG_LEN * 2 + NF * KPAD * 2 + NF * 16 * 4;   // ~42 KB dynamic LDS
  k_conv<<<dim3((NT + 15) / 16, NB), 256, ldsB, stream>>>(x, filt, win, frames);
  k_ctrl<<<(NB * NF * NT) / 256, 256, 0, stream>>>(frames, wih, whh, bih, bhh,
                                                   hw1, hb1, hw2, hb2, al, rr);
  k_pcen<<<(NB * NF + 63) / 64, 64, 0, stream>>>(frames, al, rr, out);
}
